// ConvergedToeplitzInhibition_36017595744316
// MI455X (gfx1250) — matrix operand order repaired and checked by reading, not compiled, not run on hardware
//
#include <hip/hip_runtime.h>
#include <hip/hip_bf16.h>

#define C_DIM 256
#define HW    4096   // 64*64
#define N_IMG 32
#define SCOPE 27
#define USTRIDE 132  // 128 uint k-pairs + pad; *4B keeps 16B alignment for b128

typedef __attribute__((ext_vector_type(16))) __bf16 v16bf;
typedef __attribute__((ext_vector_type(8)))  float  v8f;

// ---------------------------------------------------------------------------
// Kernel 1: build padded/rolled ricker kernel, solve (I - T) u = e0 by
// Neumann iteration entirely in LDS (circulant => u gives the full inverse).
// ---------------------------------------------------------------------------
__global__ __launch_bounds__(256)
void prep_inverse(const float* __restrict__ filt, float* __restrict__ u_out) {
    __shared__ float kv[C_DIM];
    __shared__ float v[C_DIM];
    __shared__ float vn[C_DIM];
    const int t = threadIdx.x;

    // _pad_roll: padded[114..140] = filt[0..26]; rolled[i] = padded[(i-129) mod 256]
    const int pos = (t + C_DIM - 129) & (C_DIM - 1);
    kv[t] = (pos >= 114 && pos < 114 + SCOPE) ? filt[pos - 114] : 0.0f;
    v[t]  = (t == 0) ? 1.0f : 0.0f;
    __syncthreads();

    // u_{m+1} = e0 + T u_m,  T[i][j] = kv[(j-i) mod 256]
    for (int it = 0; it < 96; ++it) {
        float s = (t == 0) ? 1.0f : 0.0f;
        #pragma unroll 8
        for (int j = 0; j < C_DIM; ++j)
            s += kv[(j - t) & (C_DIM - 1)] * v[j];
        vn[t] = s;
        __syncthreads();
        v[t] = vn[t];
        __syncthreads();
    }
    u_out[t] = v[t];
}

// ---------------------------------------------------------------------------
// Kernel 2: expand u into A-operand (M^T) fragments, pre-swizzled into the
// CDNA5 16-bit A-matrix 16x32 per-lane layout, split into bf16 hi/lo.
//   A[c][k] = M[k][c] = u[(k - c) mod 256]
//   frag index = ((ct*8 + kt)*32 + lane)*16 + e
// ---------------------------------------------------------------------------
__global__ __launch_bounds__(256)
void expand_frags(const float* __restrict__ u,
                  unsigned short* __restrict__ mt_hi,
                  unsigned short* __restrict__ mt_lo) {
    const int idx  = blockIdx.x * 256 + threadIdx.x;   // 65536 total
    const int e    = idx & 15;
    const int lane = (idx >> 4) & 31;
    const int kt   = (idx >> 9) & 7;
    const int ct   = idx >> 12;

    const int vh  = e >> 1;
    const int odd = e & 1;
    const int kk  = (vh < 4 ? 2 * vh : 16 + 2 * (vh - 4)) + ((lane >> 4) << 3) + odd;
    const int k   = kt * 32 + kk;
    const int c   = ct * 16 + (lane & 15);

    const float val = u[(k - c) & (C_DIM - 1)];
    const unsigned int hb = __float_as_uint(val) & 0xffff0000u;
    const float lof = val - __uint_as_float(hb);
    mt_hi[idx] = (unsigned short)(hb >> 16);
    mt_lo[idx] = (unsigned short)(__float_as_uint(lof) >> 16);
}

// ---------------------------------------------------------------------------
// Kernel 3: the GEMM.  Out[n,c,hw] = sum_k M[k][c] * Act[n,k,hw].
// Block = 256 threads (8 wave32) handles 16 spatial columns x 256 channels.
// Activations are split into bf16 hi/lo ONCE at staging time and stored in
// LDS as packed k-pairs, so the compute loop is pure ds_load_b128 + WMMA.
// ---------------------------------------------------------------------------
__global__ __launch_bounds__(256)
void toeplitz_gemm(const float* __restrict__ act,
                   const v16bf* __restrict__ mt_hi,
                   const v16bf* __restrict__ mt_lo,
                   float* __restrict__ out) {
    __shared__ unsigned int sHi[16 * USTRIDE];   // [p][k-pair] bf16-hi pairs
    __shared__ unsigned int sLo[16 * USTRIDE];   // [p][k-pair] bf16-lo pairs

    const int tid  = threadIdx.x;
    const int lane = tid & 31;
    const int wave = tid >> 5;
    const int n    = blockIdx.x >> 8;          // 256 blocks per image
    const int hw0  = (blockIdx.x & 255) << 4;  // 16 columns per block

    const float* abase = act + ((size_t)n * C_DIM) * HW + hw0;

    // Stage 256(k) x 16(p) tile: load f32 rows k-pairwise, split to bf16
    // hi/lo, pack pairs (k, k+1) into uints, store transposed [p][k/2].
    {
        const int q = (tid & 3) << 2;
        #pragma unroll
        for (int i = 0; i < 2; ++i) {
            const int kp = (tid >> 2) + (i << 6);   // k-pair index 0..127
            const float4 va = *(const float4*)(abase + (size_t)(2 * kp) * HW + q);
            const float4 vb = *(const float4*)(abase + (size_t)(2 * kp + 1) * HW + q);
            const float fa[4] = {va.x, va.y, va.z, va.w};
            const float fb[4] = {vb.x, vb.y, vb.z, vb.w};
            #pragma unroll
            for (int j = 0; j < 4; ++j) {
                const unsigned int ha = __float_as_uint(fa[j]) & 0xffff0000u;
                const unsigned int hb = __float_as_uint(fb[j]) & 0xffff0000u;
                const float la = fa[j] - __uint_as_float(ha);
                const float lb = fb[j] - __uint_as_float(hb);
                sHi[(q + j) * USTRIDE + kp] = hb | (ha >> 16);
                sLo[(q + j) * USTRIDE + kp] = (__float_as_uint(lb) & 0xffff0000u)
                                            | (__float_as_uint(la) >> 16);
            }
        }
    }
    __syncthreads();

    const int ct0 = wave << 1;       // two 16-row C tiles per wave
    const int ct1 = ct0 + 1;
    v8f acc0 = {0.f, 0.f, 0.f, 0.f, 0.f, 0.f, 0.f, 0.f};
    v8f acc1 = {0.f, 0.f, 0.f, 0.f, 0.f, 0.f, 0.f, 0.f};

    // B fragment (16-bit B 32x16): lanes 0-15 -> K 0..7,16..23, lanes 16-31 ->
    // K 8..15,24..31 of the k-step; column N = lane & 15.  2x4 uints.
    const int pcol  = lane & 15;
    const int ubase = pcol * USTRIDE + ((lane >> 4) << 2);

    __builtin_prefetch(&mt_hi[(ct0 * 8) * 32 + lane], 0, 3);
    __builtin_prefetch(&mt_lo[(ct0 * 8) * 32 + lane], 0, 3);

    #pragma unroll
    for (int kt = 0; kt < 8; ++kt) {
        // A fragments: 32B coalesced loads (pre-swizzled, L2-resident)
        const v16bf a0h = mt_hi[(ct0 * 8 + kt) * 32 + lane];
        const v16bf a0l = mt_lo[(ct0 * 8 + kt) * 32 + lane];
        const v16bf a1h = mt_hi[(ct1 * 8 + kt) * 32 + lane];
        const v16bf a1l = mt_lo[(ct1 * 8 + kt) * 32 + lane];

        // B fragments: 2x ds_load_b128 each, no repacking
        union { uint4 q[2]; v16bf v; } bh, bl;
        bh.q[0] = *(const uint4*)&sHi[ubase + kt * 16];
        bh.q[1] = *(const uint4*)&sHi[ubase + kt * 16 + 8];
        bl.q[0] = *(const uint4*)&sLo[ubase + kt * 16];
        bl.q[1] = *(const uint4*)&sLo[ubase + kt * 16 + 8];

        // 3-product bf16 emulation of f32: hi*hi + hi*lo + lo*hi
        acc0 = __builtin_amdgcn_wmma_f32_16x16x32_bf16(false, a0h, false, bh.v, (short)0, acc0, false, false);
        acc0 = __builtin_amdgcn_wmma_f32_16x16x32_bf16(false, a0h, false, bl.v, (short)0, acc0, false, false);
        acc0 = __builtin_amdgcn_wmma_f32_16x16x32_bf16(false, a0l, false, bh.v, (short)0, acc0, false, false);
        acc1 = __builtin_amdgcn_wmma_f32_16x16x32_bf16(false, a1h, false, bh.v, (short)0, acc1, false, false);
        acc1 = __builtin_amdgcn_wmma_f32_16x16x32_bf16(false, a1h, false, bl.v, (short)0, acc1, false, false);
        acc1 = __builtin_amdgcn_wmma_f32_16x16x32_bf16(false, a1l, false, bh.v, (short)0, acc1, false, false);
    }

    // C/D layout: VGPR r, lane -> N = lane&15, M = r + (lane<16 ? 0 : 8)
    const int mrow = (lane >> 4) << 3;
    float* obase = out + ((size_t)n * C_DIM) * HW + hw0 + pcol;
    #pragma unroll
    for (int r = 0; r < 8; ++r) {
        obase[(size_t)(ct0 * 16 + mrow + r) * HW] = acc0[r];
        obase[(size_t)(ct1 * 16 + mrow + r) * HW] = acc1[r];
    }
}

// ---------------------------------------------------------------------------
extern "C" void kernel_launch(void* const* d_in, const int* in_sizes, int n_in,
                              void* d_out, int out_size, void* d_ws, size_t ws_size,
                              hipStream_t stream) {
    const float* act  = (const float*)d_in[0];   // [32,256,64,64] f32
    const float* filt = (const float*)d_in[1];   // [27] f32
    float* out = (float*)d_out;

    char* ws = (char*)d_ws;
    float* u = (float*)ws;                                         // 1 KB
    unsigned short* mt_hi = (unsigned short*)(ws + 1024);          // 128 KB
    unsigned short* mt_lo = (unsigned short*)(ws + 1024 + C_DIM * C_DIM * 2);

    prep_inverse<<<1, 256, 0, stream>>>(filt, u);
    expand_frags<<<(C_DIM * C_DIM) / 256, 256, 0, stream>>>(u, mt_hi, mt_lo);

    const int num_blocks = N_IMG * (HW / 16);    // 8192
    toeplitz_gemm<<<num_blocks, 256, 0, stream>>>(
        act, (const v16bf*)mt_hi, (const v16bf*)mt_lo, out);
}
